// MultiHeadSelfAttention_58823872086658
// MI455X (gfx1250) — matrix operand order repaired and checked by reading, not compiled, not run on hardware
//
#include <hip/hip_runtime.h>
#include <math.h>

typedef __attribute__((ext_vector_type(16))) __bf16 v16bf;
typedef __attribute__((ext_vector_type(8)))  float  v8f;

union Frag32 { uint4 u[2]; v16bf v; };

#define T_DIM 1024
#define B_DIM 8
#define D_DIM 1024
#define H_DIM 16
#define HD    64
#define M_DIM (T_DIM * B_DIM)   // 8192

static __device__ __forceinline__ v8f wmma_bf16(v16bf a, v16bf b, v8f c) {
  return __builtin_amdgcn_wmma_f32_16x16x32_bf16(false, a, false, b, (short)0, c,
                                                 false, false);
}

// LDS byte offset (what GLOBAL_LOAD_ASYNC_TO_LDS wants in VDST: HW adds LDS_BASE)
static __device__ __forceinline__ unsigned int lds_off(const void* p) {
  return (unsigned int)(size_t)(__attribute__((address_space(3))) const char*)p;
}

// async copy 16B global -> LDS, tracked by ASYNCcnt (no VGPR staging)
static __device__ __forceinline__ void async_b128(unsigned int lds,
                                                  const void* gptr) {
  asm volatile("global_load_async_to_lds_b128 %0, %1, off"
               :: "v"(lds), "v"((unsigned long long)(size_t)gptr)
               : "memory");
}

// ---------------------------------------------------------------- f32 -> bf16
__global__ __launch_bounds__(256) void cvt_f32_bf16(const float* __restrict__ s,
                                                    __bf16* __restrict__ d, int n) {
  int i = (blockIdx.x * blockDim.x + threadIdx.x) * 4;
  if (i >= n) return;
  float4 f = *(const float4*)(s + i);
  union { __bf16 h[4]; uint2 u; } o;
  o.h[0] = (__bf16)f.x; o.h[1] = (__bf16)f.y;
  o.h[2] = (__bf16)f.z; o.h[3] = (__bf16)f.w;
  *(uint2*)(d + i) = o.u;
}

// ------------------------------------------------------------- QKV projection
// C[m,n] = sum_k A[m,k] * W[n,k]  (NT GEMM), scatter to [B,H,T,64] bf16.
__global__ __launch_bounds__(256) void gemm_qkv(
    const __bf16* __restrict__ Xb,
    const __bf16* __restrict__ Wqb, const __bf16* __restrict__ Wkb,
    const __bf16* __restrict__ Wvb,
    const float* __restrict__ bq, const float* __restrict__ bk,
    const float* __restrict__ bv,
    __bf16* __restrict__ Qg, __bf16* __restrict__ Kg, __bf16* __restrict__ Vg) {
  __shared__ __bf16 Xs[2][128][40];
  __shared__ __bf16 Ws[2][128][40];

  const int z = blockIdx.z;
  const __bf16* Wsrc = (z == 0) ? Wqb : (z == 1) ? Wkb : Wvb;
  const float*  bias = (z == 0) ? bq  : (z == 1) ? bk  : bv;
  __bf16*       dst  = (z == 0) ? Qg  : (z == 1) ? Kg  : Vg;

  const int tid  = threadIdx.x;
  const int l    = tid & 31;
  const int wid  = tid >> 5;
  const int wm   = wid & 1;        // 0..1 -> 64-row slab
  const int wn   = wid >> 1;       // 0..3 -> 32-col slab
  const int m0   = blockIdx.y * 128;
  const int n0   = blockIdx.x * 128;
  const int lrow = l & 15;
  const int kbA  = (l < 16) ? 0 : 8;
  const int kbB  = (l < 16) ? 0 : 8;
  const int ldr  = tid >> 1;
  const int lseg = (tid & 1) * 16;

  v8f acc[4][2] = {};

  // 4 async 16B copies per thread per k-tile: X row chunk + W row chunk
  auto issue_tile = [&](int k0, int buf) {
    const __bf16* gx = Xb + (size_t)(m0 + ldr) * D_DIM + k0 + lseg;
    async_b128(lds_off(&Xs[buf][ldr][lseg]),     gx);
    async_b128(lds_off(&Xs[buf][ldr][lseg + 8]), gx + 8);
    const __bf16* gw = Wsrc + (size_t)(n0 + ldr) * D_DIM + k0 + lseg;
    async_b128(lds_off(&Ws[buf][ldr][lseg]),     gw);
    async_b128(lds_off(&Ws[buf][ldr][lseg + 8]), gw + 8);
  };

  issue_tile(0, 0);
  int cur = 0;

  for (int k0 = 0; k0 < D_DIM; k0 += 32) {
    if (k0 + 32 < D_DIM) {
      issue_tile(k0 + 32, cur ^ 1);
      // 4 copies just issued; async loads complete in order per wave,
      // so <=4 outstanding means the current tile's copies have landed.
      asm volatile("s_wait_asynccnt 0x4" ::: "memory");
    } else {
      asm volatile("s_wait_asynccnt 0x0" ::: "memory");
    }
    __syncthreads();

    Frag32 a[4], b[2];
#pragma unroll
    for (int mi = 0; mi < 4; mi++) {
      const __bf16* p = &Xs[cur][wm * 64 + mi * 16 + lrow][kbA];
      a[mi].u[0] = *(const uint4*)p;
      a[mi].u[1] = *(const uint4*)(p + 16);
    }
#pragma unroll
    for (int ni = 0; ni < 2; ni++) {
      const __bf16* p = &Ws[cur][wn * 32 + ni * 16 + lrow][kbB];
      b[ni].u[0] = *(const uint4*)p;
      b[ni].u[1] = *(const uint4*)(p + 16);
    }
#pragma unroll
    for (int mi = 0; mi < 4; mi++)
#pragma unroll
      for (int ni = 0; ni < 2; ni++)
        acc[mi][ni] = wmma_bf16(a[mi].v, b[ni].v, acc[mi][ni]);

    __syncthreads();
    cur ^= 1;
  }

  // epilogue: scatter into [B,H,T,64]
  const int rhi = (l & 16) ? 8 : 0;
#pragma unroll
  for (int ni = 0; ni < 2; ni++) {
    const int n  = n0 + wn * 32 + ni * 16 + lrow;
    const float bn = bias[n];
    const int h  = n >> 6;
    const int dd = n & 63;
#pragma unroll
    for (int mi = 0; mi < 4; mi++) {
      const int mbase = m0 + wm * 64 + mi * 16 + rhi;
#pragma unroll
      for (int r = 0; r < 8; r++) {
        const int m = mbase + r;
        const int t = m >> 3;       // m = t*B + b, B=8
        const int bb = m & 7;
        float val = acc[mi][ni][r] + bn;
        dst[((size_t)(bb * H_DIM + h) * T_DIM + t) * HD + dd] = (__bf16)val;
      }
    }
  }
}

// ------------------------------------------------------------ flash attention
__global__ __launch_bounds__(256) void attn(
    const __bf16* __restrict__ Qg, const __bf16* __restrict__ Kg,
    const __bf16* __restrict__ Vg, __bf16* __restrict__ Ab) {
  __shared__ __bf16 Ks[64][72];
  __shared__ __bf16 Vt[64][72];
  __shared__ __bf16 Pl[8][16][72];

  const int tid  = threadIdx.x;
  const int l    = tid & 31;
  const int wid  = tid >> 5;
  const int lrow = l & 15;
  const int kbA  = (l < 16) ? 0 : 8;
  const int kbB  = (l < 16) ? 0 : 8;
  const int bh   = blockIdx.y;                  // b*H + h
  const int b    = bh >> 4;
  const int h    = bh & 15;
  const size_t head_base = (size_t)bh * T_DIM * HD;
  const int q0   = blockIdx.x * 128 + wid * 16; // first query row of this wave

  // Q fragments (loop invariant): A-layout, d=64 -> two 16x32 k-steps
  Frag32 qf[2];
  {
    const __bf16* qp = Qg + head_base + (size_t)(q0 + lrow) * HD;
#pragma unroll
    for (int ks = 0; ks < 2; ks++) {
      qf[ks].u[0] = *(const uint4*)(qp + ks * 32 + kbA);
      qf[ks].u[1] = *(const uint4*)(qp + ks * 32 + kbA + 16);
    }
  }

  float mrun[8], srun[8];
#pragma unroll
  for (int r = 0; r < 8; r++) { mrun[r] = -3.0e38f; srun[r] = 0.0f; }
  v8f O[4] = {};
  const float scale = 0.125f;  // 1/sqrt(64)

  const int li   = tid >> 2;
  const int vseg = (tid & 3) * 16;

  for (int c = 0; c < T_DIM; c += 64) {
    __syncthreads();
    {  // K chunk: async global->LDS (row-major); V chunk: register transpose
      const __bf16* gk = Kg + head_base + (size_t)(c + li) * HD + vseg;
      async_b128(lds_off(&Ks[li][vseg]),     gk);
      async_b128(lds_off(&Ks[li][vseg + 8]), gk + 8);

      Frag32 fv;
      const uint4* pv = (const uint4*)(Vg + head_base + (size_t)(c + li) * HD + vseg);
      fv.u[0] = pv[0]; fv.u[1] = pv[1];
#pragma unroll
      for (int q = 0; q < 16; q++) Vt[vseg + q][li] = fv.v[q];
    }
    asm volatile("s_wait_asynccnt 0x0" ::: "memory");
    __syncthreads();

    // scores S = Q * K^T for 4 key subtiles of 16
    v8f S[4];
#pragma unroll
    for (int ni = 0; ni < 4; ni++) {
      v8f accs = {};
#pragma unroll
      for (int ks = 0; ks < 2; ks++) {
        Frag32 kf;
        const __bf16* p = &Ks[ni * 16 + lrow][ks * 32 + kbB];
        kf.u[0] = *(const uint4*)p;
        kf.u[1] = *(const uint4*)(p + 16);
        accs = wmma_bf16(qf[ks].v, kf.v, accs);
      }
      S[ni] = accs;
    }

    // flash running-max update (row reduce across 16-lane C-layout groups)
    float alpha[8];
#pragma unroll
    for (int r = 0; r < 8; r++) {
      float cm = fmaxf(fmaxf(S[0][r], S[1][r]), fmaxf(S[2][r], S[3][r])) * scale;
      cm = fmaxf(cm, __shfl_xor(cm, 1));
      cm = fmaxf(cm, __shfl_xor(cm, 2));
      cm = fmaxf(cm, __shfl_xor(cm, 4));
      cm = fmaxf(cm, __shfl_xor(cm, 8));
      float mnew = fmaxf(mrun[r], cm);
      alpha[r] = __expf(mrun[r] - mnew);
      mrun[r] = mnew;
    }

    // P = exp(S*scale - m), rowsum, and bounce P through LDS -> A layout
    const int prow = (l & 16) ? 8 : 0;
    float rs[8];
#pragma unroll
    for (int r = 0; r < 8; r++) rs[r] = 0.0f;
#pragma unroll
    for (int ni = 0; ni < 4; ni++) {
#pragma unroll
      for (int r = 0; r < 8; r++) {
        float pv = __expf(S[ni][r] * scale - mrun[r]);
        rs[r] += pv;
        Pl[wid][prow + r][ni * 16 + lrow] = (__bf16)pv;
      }
    }
#pragma unroll
    for (int r = 0; r < 8; r++) {
      float s = rs[r];
      s += __shfl_xor(s, 1); s += __shfl_xor(s, 2);
      s += __shfl_xor(s, 4); s += __shfl_xor(s, 8);
      srun[r] = srun[r] * alpha[r] + s;
    }
#pragma unroll
    for (int ni = 0; ni < 4; ni++)
#pragma unroll
      for (int r = 0; r < 8; r++) O[ni][r] *= alpha[r];

    asm volatile("" ::: "memory");  // keep P stores before P reloads

    // O += P * V  (same-wave DS ops are hardware in-order)
#pragma unroll
    for (int jb = 0; jb < 2; jb++) {
      Frag32 pf;
      const __bf16* pp = &Pl[wid][lrow][jb * 32 + kbA];
      pf.u[0] = *(const uint4*)pp;
      pf.u[1] = *(const uint4*)(pp + 16);
#pragma unroll
      for (int ni = 0; ni < 4; ni++) {
        Frag32 vf;
        const __bf16* vp = &Vt[ni * 16 + lrow][jb * 32 + kbB];
        vf.u[0] = *(const uint4*)vp;
        vf.u[1] = *(const uint4*)(vp + 16);
        O[ni] = wmma_bf16(pf.v, vf.v, O[ni]);
      }
    }
  }

  // epilogue: normalize and write [T*B, D] bf16 for the output projection
  const int rhi = (l & 16) ? 8 : 0;
#pragma unroll
  for (int ni = 0; ni < 4; ni++) {
    const int n = h * 64 + ni * 16 + lrow;
#pragma unroll
    for (int r = 0; r < 8; r++) {
      const int t = q0 + rhi + r;
      float val = O[ni][r] / srun[r];
      Ab[(size_t)(t * B_DIM + b) * D_DIM + n] = (__bf16)val;
    }
  }
}

// ----------------------------------------------------------- output projection
__global__ __launch_bounds__(256) void gemm_out(
    const __bf16* __restrict__ Abf, const __bf16* __restrict__ Wob,
    const float* __restrict__ bo, float* __restrict__ out) {
  __shared__ __bf16 Xs[2][128][40];
  __shared__ __bf16 Ws[2][128][40];

  const int tid  = threadIdx.x;
  const int l    = tid & 31;
  const int wid  = tid >> 5;
  const int wm   = wid & 1;
  const int wn   = wid >> 1;
  const int m0   = blockIdx.y * 128;
  const int n0   = blockIdx.x * 128;
  const int lrow = l & 15;
  const int kbA  = (l < 16) ? 0 : 8;
  const int kbB  = (l < 16) ? 0 : 8;
  const int ldr  = tid >> 1;
  const int lseg = (tid & 1) * 16;

  v8f acc[4][2] = {};

  auto issue_tile = [&](int k0, int buf) {
    const __bf16* gx = Abf + (size_t)(m0 + ldr) * D_DIM + k0 + lseg;
    async_b128(lds_off(&Xs[buf][ldr][lseg]),     gx);
    async_b128(lds_off(&Xs[buf][ldr][lseg + 8]), gx + 8);
    const __bf16* gw = Wob + (size_t)(n0 + ldr) * D_DIM + k0 + lseg;
    async_b128(lds_off(&Ws[buf][ldr][lseg]),     gw);
    async_b128(lds_off(&Ws[buf][ldr][lseg + 8]), gw + 8);
  };

  issue_tile(0, 0);
  int cur = 0;

  for (int k0 = 0; k0 < D_DIM; k0 += 32) {
    if (k0 + 32 < D_DIM) {
      issue_tile(k0 + 32, cur ^ 1);
      asm volatile("s_wait_asynccnt 0x4" ::: "memory");
    } else {
      asm volatile("s_wait_asynccnt 0x0" ::: "memory");
    }
    __syncthreads();

    Frag32 a[4], b[2];
#pragma unroll
    for (int mi = 0; mi < 4; mi++) {
      const __bf16* p = &Xs[cur][wm * 64 + mi * 16 + lrow][kbA];
      a[mi].u[0] = *(const uint4*)p;
      a[mi].u[1] = *(const uint4*)(p + 16);
    }
#pragma unroll
    for (int ni = 0; ni < 2; ni++) {
      const __bf16* p = &Ws[cur][wn * 32 + ni * 16 + lrow][kbB];
      b[ni].u[0] = *(const uint4*)p;
      b[ni].u[1] = *(const uint4*)(p + 16);
    }
#pragma unroll
    for (int mi = 0; mi < 4; mi++)
#pragma unroll
      for (int ni = 0; ni < 2; ni++)
        acc[mi][ni] = wmma_bf16(a[mi].v, b[ni].v, acc[mi][ni]);

    __syncthreads();
    cur ^= 1;
  }

  const int rhi = (l & 16) ? 8 : 0;
#pragma unroll
  for (int ni = 0; ni < 2; ni++) {
    const int n = n0 + wn * 32 + ni * 16 + lrow;
    const float bn = bo[n];
#pragma unroll
    for (int mi = 0; mi < 4; mi++) {
      const int mbase = m0 + wm * 64 + mi * 16 + rhi;
#pragma unroll
      for (int r = 0; r < 8; r++) {
        __builtin_nontemporal_store(acc[mi][ni][r] + bn,
                                    out + (size_t)(mbase + r) * D_DIM + n);
      }
    }
  }
}

// ---------------------------------------------------------------------- host
extern "C" void kernel_launch(void* const* d_in, const int* in_sizes, int n_in,
                              void* d_out, int out_size, void* d_ws, size_t ws_size,
                              hipStream_t stream) {
  (void)in_sizes; (void)n_in; (void)out_size; (void)ws_size;
  const float* x  = (const float*)d_in[0];
  const float* Wq = (const float*)d_in[1];
  const float* bq = (const float*)d_in[2];
  const float* Wk = (const float*)d_in[3];
  const float* bk = (const float*)d_in[4];
  const float* Wv = (const float*)d_in[5];
  const float* bv = (const float*)d_in[6];
  const float* Wo = (const float*)d_in[7];
  const float* bo = (const float*)d_in[8];
  float* out = (float*)d_out;

  const size_t MD = (size_t)M_DIM * D_DIM;   // 8388608
  const size_t DD = (size_t)D_DIM * D_DIM;   // 1048576
  char* ws = (char*)d_ws;
  size_t off = 0;
  __bf16* Xb  = (__bf16*)(ws + off); off += MD * 2;
  __bf16* Wqb = (__bf16*)(ws + off); off += DD * 2;
  __bf16* Wkb = (__bf16*)(ws + off); off += DD * 2;
  __bf16* Wvb = (__bf16*)(ws + off); off += DD * 2;
  __bf16* Wob = (__bf16*)(ws + off); off += DD * 2;
  __bf16* Qg  = (__bf16*)(ws + off); off += MD * 2;
  __bf16* Kg  = (__bf16*)(ws + off); off += MD * 2;
  __bf16* Vg  = (__bf16*)(ws + off); off += MD * 2;
  __bf16* Ab  = (__bf16*)(ws + off); off += MD * 2;

  cvt_f32_bf16<<<(int)(MD / 4 + 255) / 256, 256, 0, stream>>>(x, Xb, (int)MD);
  cvt_f32_bf16<<<(int)(DD / 4 + 255) / 256, 256, 0, stream>>>(Wq, Wqb, (int)DD);
  cvt_f32_bf16<<<(int)(DD / 4 + 255) / 256, 256, 0, stream>>>(Wk, Wkb, (int)DD);
  cvt_f32_bf16<<<(int)(DD / 4 + 255) / 256, 256, 0, stream>>>(Wv, Wvb, (int)DD);
  cvt_f32_bf16<<<(int)(DD / 4 + 255) / 256, 256, 0, stream>>>(Wo, Wob, (int)DD);

  gemm_qkv<<<dim3(D_DIM / 128, M_DIM / 128, 3), 256, 0, stream>>>(
      Xb, Wqb, Wkb, Wvb, bq, bk, bv, Qg, Kg, Vg);

  attn<<<dim3(T_DIM / 128, B_DIM * H_DIM), 256, 0, stream>>>(Qg, Kg, Vg, Ab);

  gemm_out<<<dim3(D_DIM / 128, M_DIM / 128), 256, 0, stream>>>(Ab, Wob, bo, out);
}
